// MultiHeadedAttention_5282809774915
// MI455X (gfx1250) — matrix operand order repaired and checked by reading, not compiled, not run on hardware
//
#include <hip/hip_runtime.h>

// ---------------------------------------------------------------------------
// MI455X (gfx1250) attention: bf16 WMMA GEMMs + RoPE + softmax.
// Compute-bound (~275 GFLOP vs ~400MB traffic) -> v_wmma_f32_16x16x32_bf16.
// Data movement: GLOBAL_LOAD_ASYNC_TO_LDS_B128 (ASYNCcnt) + double-buffered LDS.
// wave32; 256-thread blocks = 8 waves; 128x256 block tile; 64x64 per wave.
// ---------------------------------------------------------------------------

typedef __attribute__((ext_vector_type(16))) __bf16 v16bf;
typedef __attribute__((ext_vector_type(8)))  float  v8f;
typedef int v4i __attribute__((vector_size(16)));            // matches builtin pointee
typedef __attribute__((address_space(1))) v4i* gv4i_p;       // global (AS1) int4*
typedef __attribute__((address_space(3))) v4i* lv4i_p;       // LDS (AS3) int4*

union Frag32B { uint4 u4[2]; v16bf v; };   // 32 bytes = one WMMA A/B fragment per lane

#define TB 256
#define BM 128
#define BN 256
#define BK 32
#define LDSW 40   // halves per LDS row: 32 data + 8 pad (16B) -> 80B stride, 16B aligned

// ---- CDNA5 async global->LDS (no VGPR round trip, tracked by ASYNCcnt) ----
#if __has_builtin(__builtin_amdgcn_global_load_async_to_lds_b128)
__device__ __forceinline__ void async_copy_b128(void* lds_dst, const void* gsrc) {
  __builtin_amdgcn_global_load_async_to_lds_b128((gv4i_p)gsrc, (lv4i_p)lds_dst, 0, 0);
}
#else
__device__ __forceinline__ void async_copy_b128(void* lds_dst, const void* gsrc) {
  unsigned loff = (unsigned)(size_t)(__attribute__((address_space(3))) void*)lds_dst;
  asm volatile("global_load_async_to_lds_b128 %0, %1, off"
               :: "v"(loff), "v"(gsrc) : "memory");
}
#endif

__device__ __forceinline__ void wait_async_all() {
#if __has_builtin(__builtin_amdgcn_s_wait_asynccnt)
  __builtin_amdgcn_s_wait_asynccnt(0);
#else
  asm volatile("s_wait_asynccnt 0x0" ::: "memory");
#endif
}

// ---------------- f32 -> bf16 bulk convert (vectorized) ----------------
__global__ __launch_bounds__(TB)
void cvt_f32_bf16(const float* __restrict__ in, __bf16* __restrict__ out, int n4) {
  int i = blockIdx.x * blockDim.x + threadIdx.x;
  if (i < n4) {
    float4 f = ((const float4*)in)[i];
    union { __bf16 b[4]; ushort4 u; } p;
    p.b[0] = (__bf16)f.x; p.b[1] = (__bf16)f.y;
    p.b[2] = (__bf16)f.z; p.b[3] = (__bf16)f.w;
    ((ushort4*)out)[i] = p.u;
  }
}

// ---------------- tiled WMMA GEMM: C = alpha * A * B^T ----------------
// A: M x K bf16 row-major.  B: N x K bf16 row-major (computes A @ B.T).
// SMODE 0: C f32 [m*N+n]   1: C bf16 [m*N+n]   2: C bf16 transposed [n*M+m]
template<int SMODE>
__global__ __launch_bounds__(TB)
void gemm_bf16_wmma(const __bf16* __restrict__ A, const __bf16* __restrict__ B,
                    void* __restrict__ Cv, int M, int N, int K, float alpha) {
  __shared__ alignas(16) __bf16 sA[2][BM][LDSW];   // 20 KB
  __shared__ alignas(16) __bf16 sB[2][BN][LDSW];   // 40 KB

  const int tid   = threadIdx.x;
  const int wave  = tid >> 5;
  const int lane  = tid & 31;
  const int warpM = wave >> 2;            // 0..1  (64-row slab)
  const int warpN = wave & 3;             // 0..3  (64-col slab)
  const int mbase = blockIdx.y * BM;
  const int nbase = blockIdx.x * BN;
  const int r16   = lane & 15;
  const int hi16  = lane >> 4;

  v8f acc[4][4];
  #pragma unroll
  for (int i = 0; i < 4; ++i)
    #pragma unroll
    for (int j = 0; j < 4; ++j) {
      v8f z = {0.f,0.f,0.f,0.f,0.f,0.f,0.f,0.f};
      acc[i][j] = z;
    }

  // One K-tile: A 128x64B = 512 x16B chunks (2/thread), B 256x64B = 1024 (4/thread)
  auto load_tile = [&](int t, int buf) {
    const int k0 = t * BK;
    #pragma unroll
    for (int i = 0; i < 2; ++i) {
      int c = tid + i * TB, row = c >> 2, seg = c & 3;
      async_copy_b128(&sA[buf][row][seg * 8],
                      A + (size_t)(mbase + row) * K + k0 + seg * 8);
    }
    #pragma unroll
    for (int i = 0; i < 4; ++i) {
      int c = tid + i * TB, row = c >> 2, seg = c & 3;
      async_copy_b128(&sB[buf][row][seg * 8],
                      B + (size_t)(nbase + row) * K + k0 + seg * 8);
    }
  };

  const int TILES = K / BK;
  load_tile(0, 0);                         // prologue: stream tile 0 into buf 0

  for (int t = 0; t < TILES; ++t) {
    const int cur = t & 1;
    wait_async_all();                      // my tile-t async writes landed in LDS
    __syncthreads();                       // everyone's landed; other buf free
    if (t + 1 < TILES) load_tile(t + 1, cur ^ 1);   // overlap with compute below

    // Fragments per ISA 7.12.2 (16-bit A 16x32 / B 32x16 layouts)
    Frag32B afr[4], bfr[4];
    #pragma unroll
    for (int i = 0; i < 4; ++i) {
      int row  = warpM * 64 + i * 16 + r16;
      int koff = hi16 * 8;                 // lanes 16-31 hold K=8..15 / 24..31
      afr[i].u4[0] = *(const uint4*)&sA[cur][row][koff];
      afr[i].u4[1] = *(const uint4*)&sA[cur][row][koff + 16];
    }
    #pragma unroll
    for (int j = 0; j < 4; ++j) {
      int col  = warpN * 64 + j * 16 + r16;
      int koff = hi16 * 8;                 // lanes 16-31 hold K=8..15 / 24..31
      bfr[j].u4[0] = *(const uint4*)&sB[cur][col][koff];
      bfr[j].u4[1] = *(const uint4*)&sB[cur][col][koff + 16];
    }
    #pragma unroll
    for (int i = 0; i < 4; ++i)
      #pragma unroll
      for (int j = 0; j < 4; ++j)
        acc[i][j] = __builtin_amdgcn_wmma_f32_16x16x32_bf16(
            false, afr[i].v, false, bfr[j].v, (short)0, acc[i][j], false, false);
  }

  // Epilogue: C/D layout -> VGPR r: lanes0-15 M=r, lanes16-31 M=r+8; N=lane&15
  const int mo = hi16 * 8;
  #pragma unroll
  for (int i = 0; i < 4; ++i) {
    #pragma unroll
    for (int j = 0; j < 4; ++j) {
      int m0 = mbase + warpM * 64 + i * 16 + mo;
      int n  = nbase + warpN * 64 + j * 16 + r16;
      #pragma unroll
      for (int r = 0; r < 8; ++r) {
        float val = acc[i][j][r] * alpha;
        int m = m0 + r;
        if (SMODE == 0)      ((float *)Cv)[(size_t)m * N + n] = val;
        else if (SMODE == 1) ((__bf16*)Cv)[(size_t)m * N + n] = (__bf16)val;
        else                 ((__bf16*)Cv)[(size_t)n * M + m] = (__bf16)val;  // store C^T
      }
    }
  }
}

// ---------------- RoPE in place on bf16 (rope_dim == D, half = D/2) ----------------
__global__ __launch_bounds__(TB)
void rope_inplace_bf16(__bf16* __restrict__ t, int D, int half) {
  int idx = blockIdx.x * blockDim.x + threadIdx.x;   // S*half threads
  int s = idx / half;
  int i = idx - s * half;
  float inv = __powf(10000.0f, -(float)i / (float)half);
  float ang = (float)s * inv;
  float c = cosf(ang), sn = sinf(ang);
  size_t base = (size_t)s * D;
  float x1 = (float)t[base + i];
  float x2 = (float)t[base + i + half];
  t[base + i]        = (__bf16)(x1 * c - x2 * sn);
  t[base + i + half] = (__bf16)(x2 * c + x1 * sn);
}

// ---------------- row softmax: f32 scores -> bf16 probs ----------------
__global__ __launch_bounds__(TB)
void softmax_row(const float* __restrict__ scores, __bf16* __restrict__ probs, int N) {
  const int row = blockIdx.x;
  const float* x = scores + (size_t)row * N;
  __shared__ float red[TB];

  float m = -3.402823466e38f;
  for (int j = threadIdx.x; j < N; j += TB) m = fmaxf(m, x[j]);
  red[threadIdx.x] = m; __syncthreads();
  for (int s = TB / 2; s > 0; s >>= 1) {
    if (threadIdx.x < s) red[threadIdx.x] = fmaxf(red[threadIdx.x], red[threadIdx.x + s]);
    __syncthreads();
  }
  m = red[0]; __syncthreads();

  float acc = 0.f;
  for (int j = threadIdx.x; j < N; j += TB) acc += expf(x[j] - m);
  red[threadIdx.x] = acc; __syncthreads();
  for (int s = TB / 2; s > 0; s >>= 1) {
    if (threadIdx.x < s) red[threadIdx.x] += red[threadIdx.x + s];
    __syncthreads();
  }
  float inv = 1.0f / (red[0] + 1e-8f);

  __bf16* p = probs + (size_t)row * N;
  for (int j = threadIdx.x; j < N; j += TB) p[j] = (__bf16)(expf(x[j] - m) * inv);
}

// ---------------------------------------------------------------------------
extern "C" void kernel_launch(void* const* d_in, const int* in_sizes, int n_in,
                              void* d_out, int out_size, void* d_ws, size_t ws_size,
                              hipStream_t stream) {
  constexpr int S = 4096, D = 2048;
  constexpr float SCALE = 16.0f;

  const float* X  = (const float*)d_in[0];
  const float* Wq = (const float*)d_in[1];
  const float* Wk = (const float*)d_in[2];
  const float* Wv = (const float*)d_in[3];
  const float* Wo = (const float*)d_in[4];

  // workspace layout (all sizes are 256B multiples); total ~208 MiB
  char* ws = (char*)d_ws;
  size_t o = 0;
  auto carve = [&](size_t bytes) { char* p = ws + o; o += (bytes + 255) & ~(size_t)255; return p; };
  __bf16* Xb    = (__bf16*)carve((size_t)S * D * 2);
  __bf16* Wqb   = (__bf16*)carve((size_t)D * D * 2);
  __bf16* Wkb   = (__bf16*)carve((size_t)D * D * 2);
  __bf16* Wvb   = (__bf16*)carve((size_t)D * D * 2);
  __bf16* Wob   = (__bf16*)carve((size_t)D * D * 2);
  __bf16* qb    = (__bf16*)carve((size_t)S * D * 2);   // q (post-proj, roped in place)
  __bf16* kb    = (__bf16*)carve((size_t)S * D * 2);
  __bf16* vTb   = (__bf16*)carve((size_t)D * S * 2);   // v stored transposed: vT[d][s]
  __bf16* attnb = (__bf16*)carve((size_t)S * D * 2);   // probs @ v (bf16 for final GEMM)
  float*  scoresf = (float*)carve((size_t)S * S * 4);
  __bf16* probsb  = (__bf16*)carve((size_t)S * S * 2);
  (void)ws_size;

  // 1) convert inputs to bf16
  cvt_f32_bf16<<<(S * D / 4 + TB - 1) / TB, TB, 0, stream>>>(X,  Xb,  S * D / 4);
  cvt_f32_bf16<<<(D * D / 4 + TB - 1) / TB, TB, 0, stream>>>(Wq, Wqb, D * D / 4);
  cvt_f32_bf16<<<(D * D / 4 + TB - 1) / TB, TB, 0, stream>>>(Wk, Wkb, D * D / 4);
  cvt_f32_bf16<<<(D * D / 4 + TB - 1) / TB, TB, 0, stream>>>(Wv, Wvb, D * D / 4);
  cvt_f32_bf16<<<(D * D / 4 + TB - 1) / TB, TB, 0, stream>>>(Wo, Wob, D * D / 4);

  // 2) projections: q = X Wq^T, k = X Wk^T (bf16), v = X Wv^T stored transposed
  dim3 gProj(D / BN, S / BM);
  gemm_bf16_wmma<1><<<gProj, TB, 0, stream>>>(Xb, Wqb, qb,  S, D, D, 1.0f);
  gemm_bf16_wmma<1><<<gProj, TB, 0, stream>>>(Xb, Wkb, kb,  S, D, D, 1.0f);
  gemm_bf16_wmma<2><<<gProj, TB, 0, stream>>>(Xb, Wvb, vTb, S, D, D, 1.0f);

  // 3) RoPE in place on q and k (rope_dim == D)
  const int half = D / 2;
  rope_inplace_bf16<<<(S * half) / TB, TB, 0, stream>>>(qb, D, half);
  rope_inplace_bf16<<<(S * half) / TB, TB, 0, stream>>>(kb, D, half);

  // 4) scores = q k^T / SCALE  (f32 output)
  dim3 gScore(S / BN, S / BM);
  gemm_bf16_wmma<0><<<gScore, TB, 0, stream>>>(qb, kb, scoresf, S, S, D, 1.0f / SCALE);

  // 5) softmax rows -> bf16 probs
  softmax_row<<<S, TB, 0, stream>>>(scoresf, probsb, S);

  // 6) attn = probs @ v  (B = vT, row n holds v[:, n] contiguously over K=S)
  dim3 gAttn(D / BN, S / BM);
  gemm_bf16_wmma<1><<<gAttn, TB, 0, stream>>>(probsb, vTb, attnb, S, D, S, 1.0f);

  // 7) out = attn @ Wo^T  (f32 straight to d_out)
  gemm_bf16_wmma<0><<<gAttn, TB, 0, stream>>>(attnb, Wob, d_out, S, D, D, 1.0f);
  (void)in_sizes; (void)n_in; (void)out_size;
}
